// RGPNet_38783554683239
// MI455X (gfx1250) — matrix operand order repaired and checked by reading, not compiled, not run on hardware
//
#include <hip/hip_runtime.h>
#include <hip/hip_bf16.h>

// CDNA5 / gfx1250 implementation of RGPNet.
// Conv + FC layers run as implicit-GEMM on v_wmma_f32_16x16x32_f16 (wave32).
// Conv: each 256-thread block stages a zero-padded activation patch + im2col
// offset table in LDS; B fragments gather from LDS (branch-free), A fragments
// are contiguous global b128 loads from Kpad-padded f16 weights; one wave
// holds all COUT/16 accumulator tiles so each B fragment feeds COUT/16 WMMAs.
// Both GEMM inner loops are software-pipelined (operands for chunk k+1 issue
// before the WMMA of chunk k) to hide load latency behind the matrix pipe.
// Workspace requirement: ~270 MB (local branch aliases dead global buffers).

typedef __attribute__((ext_vector_type(16))) _Float16 v16h;
typedef __attribute__((ext_vector_type(8)))  _Float16 h8;
typedef __attribute__((ext_vector_type(8)))  float    v8f;

__device__ __forceinline__ v16h cat16(h8 lo, h8 hi) {
  return __builtin_shufflevector(lo, hi, 0, 1, 2, 3, 4, 5, 6, 7,
                                 8, 9, 10, 11, 12, 13, 14, 15);
}

// ---------------------------------------------------------------- utilities

__global__ void k_f32_to_f16(const float* __restrict__ src, _Float16* __restrict__ dst, int n) {
  int i = blockIdx.x * blockDim.x + threadIdx.x;
  if (i < n) dst[i] = (_Float16)src[i];
}

// pack OIHW f32 weights -> f16 [COUT][Kpad], zero-padded K..Kpad
__global__ void k_pack_w(const float* __restrict__ src, _Float16* __restrict__ dst,
                         int COUT, int K, int Kpad) {
  int i = blockIdx.x * blockDim.x + threadIdx.x;
  if (i >= COUT * Kpad) return;
  const int o = i / Kpad, k = i - o * Kpad;
  dst[i] = (k < K) ? (_Float16)src[o * K + k] : (_Float16)0.f;
}

// ------------------------------------------------------- implicit-GEMM conv
// WMMA fragment layouts per CDNA5 ISA 7.12.2 (wave32):
//   A: lane row m = lane&15; halves 0-7 hold K = kh*8+0..7 (contiguous),
//      halves 8-15 hold K = 16+kh*8+0..7 (contiguous)  -> two b128 loads.
//   B: lane col n = lane&15; halves 0-7 hold K = kh*8+0..7, 8-15 hold K = 16+kh*8+0..7 (same as A).
//   D: lane col n = lane&15; VGPR v holds row m = v + 8*kh.
template <int CIN, int COUT, int KW, int PAD, int H, int W>
__global__ __launch_bounds__(256) void k_conv_wmma(
    const _Float16* __restrict__ in,   // [nimg][CIN][H][W]
    const _Float16* __restrict__ wgt,  // [COUT][Kpad]
    _Float16* __restrict__ out) {      // [nimg][COUT][H][W]
  constexpr int K     = CIN * KW * KW;
  constexpr int Kpad  = (K + 31) & ~31;
  constexpr int HW    = H * W;
  constexpr int PB    = 128;                 // pixels per block (8 waves x 16)
  constexpr int WP    = W + 2 * PAD;         // padded patch width
  constexpr int SPAN0 = (PB + W - 2) / W + 1;
  constexpr int SPAN  = SPAN0 < H ? SPAN0 : H;
  constexpr int RS    = SPAN + 2 * PAD;      // staged rows
  constexpr int MT    = COUT / 16;           // accumulator tiles per wave

  __shared__ _Float16 patch[CIN * RS * WP];
  __shared__ int      koff[Kpad];

  const int img   = blockIdx.x;
  const int pbase = blockIdx.y * PB;
  const int tid   = threadIdx.x;
  const int ry0   = pbase / W;

  const _Float16* __restrict__ inimg = in + (size_t)img * CIN * HW;

  // ---- stage zero-padded activation patch
  for (int idx = tid; idx < CIN * RS * WP; idx += 256) {
    const int xx  = idx % WP;
    int t2        = idx / WP;
    const int rr  = t2 % RS;
    const int cin = t2 / RS;
    const int y = ry0 - PAD + rr;
    const int x = xx - PAD;
    _Float16 v = (_Float16)0.f;
    if (y >= 0 && y < H && x >= 0 && x < W) v = inimg[(size_t)cin * HW + y * W + x];
    patch[idx] = v;
  }
  // ---- stage im2col offset table (k -> LDS offset, clamped for pad region)
  for (int k = tid; k < Kpad; k += 256) {
    const int kk  = k < K ? k : K - 1;
    const int cin = kk / (KW * KW);
    const int r   = kk - cin * (KW * KW);
    const int dy  = r / KW;
    const int dx  = r - dy * KW;
    koff[k] = (cin * RS + dy) * WP + dx;
  }
  __syncthreads();

  const int wave = tid >> 5;
  const int lane = tid & 31;
  const int kh   = lane >> 4;
  const int lid  = lane & 15;
  const int nbase = pbase + wave * 16;
  if (nbase >= HW) return;  // wave-uniform: EXEC stays all-ones for WMMA

  int p = nbase + lid;
  const bool pvalid = p < HW;
  if (p >= HW) p = HW - 1;
  const int py  = p / W;
  const int pxl = p - py * W;
  const int lbase = (py - ry0) * WP + pxl;

  v8f c[MT];
  const v8f vzero = {0.f, 0.f, 0.f, 0.f, 0.f, 0.f, 0.f, 0.f};
#pragma unroll
  for (int mt = 0; mt < MT; ++mt) c[mt] = vzero;

  auto gatherB = [&](int kc) -> v16h {
    v16h b;
#pragma unroll
    for (int i = 0; i < 16; ++i)
      b[i] = patch[lbase + koff[kc + kh * 8 + i + (i >= 8 ? 8 : 0)]];
    return b;
  };

  // software-pipelined: B(kc+32) gathers issue before the WMMA chain of kc,
  // A m-tile loads are staged one tile ahead within the chain.
  v16h bnext = gatherB(0);
  for (int kc = 0; kc < Kpad; kc += 32) {
    const v16h bc = bnext;
    if (kc + 32 < Kpad) bnext = gatherB(kc + 32);
    const _Float16* wk = wgt + (size_t)lid * Kpad + kc + kh * 8;
    h8 alo = *(const h8*)(wk);
    h8 ahi = *(const h8*)(wk + 16);
#pragma unroll
    for (int mt = 0; mt < MT; ++mt) {
      const h8 clo = alo, chi = ahi;
      if (mt + 1 < MT) {
        const _Float16* wr = wk + (size_t)(mt + 1) * 16 * Kpad;
        alo = *(const h8*)(wr);
        ahi = *(const h8*)(wr + 16);
      }
      c[mt] = __builtin_amdgcn_wmma_f32_16x16x32_f16(false, cat16(clo, chi), false, bc,
                                                     (short)0, c[mt], false, false);
    }
  }
  if (pvalid) {
#pragma unroll
    for (int mt = 0; mt < MT; ++mt) {
#pragma unroll
      for (int v = 0; v < 8; ++v) {
        const int m = mt * 16 + v + 8 * kh;
        float val = c[mt][v];
        val = val > 0.f ? val : 0.01f * val;  // leaky_relu(0.01)
        out[((size_t)img * COUT + m) * HW + nbase + lid] = (_Float16)val;
      }
    }
  }
}

// --------------------------------------------------------------- 2x2 maxpool
__global__ void k_pool2(const _Float16* __restrict__ in, _Float16* __restrict__ out,
                        int C, int H, int W, int nimg) {
  const int Ho = H / 2, Wo = W / 2;
  const size_t total = (size_t)nimg * C * Ho * Wo;
  size_t i = (size_t)blockIdx.x * blockDim.x + threadIdx.x;
  if (i >= total) return;
  int wo = (int)(i % Wo); size_t t = i / Wo;
  int ho = (int)(t % Ho); t /= Ho;
  int cc = (int)(t % C);  size_t img = t / C;
  const _Float16* pI = in + ((img * C + cc) * (size_t)H + ho * 2) * W + wo * 2;
  float v0 = (float)pI[0], v1 = (float)pI[1];
  float v2 = (float)pI[W], v3 = (float)pI[W + 1];
  out[((img * C + cc) * (size_t)Ho + ho) * Wo + wo] =
      (_Float16)fmaxf(fmaxf(v0, v1), fmaxf(v2, v3));
}

// ------------------------------------------------------------ ROI crop/resize
__global__ void k_roi(const float* __restrict__ x, const float* __restrict__ px,
                      _Float16* __restrict__ crops) {
  int idx = blockIdx.x * blockDim.x + threadIdx.x;
  if (idx >= 240 * 8 * 14 * 14) return;
  int q = idx % 14; int t = idx / 14;
  int p = t % 14;   t /= 14;
  int r = t % 8;    int bt = t / 8;
  const float cx = px[(bt * 8 + r) * 2 + 0];
  const float cy = px[(bt * 8 + r) * 2 + 1];
  const float x1 = floorf(fminf(fmaxf(cx - 7.f, 0.f), 63.f));
  const float x2 = floorf(fminf(cx + 7.f, 64.f));
  const float y1 = floorf(fminf(fmaxf(cy - 7.f, 0.f), 63.f));
  const float y2 = floorf(fminf(cy + 7.f, 64.f));
  float sx = x1 + ((float)q + 0.5f) * (x2 - x1) * (1.f / 14.f) - 0.5f;
  sx = fminf(fmaxf(sx, x1), x2 - 1.f);
  float sy = y1 + ((float)p + 0.5f) * (y2 - y1) * (1.f / 14.f) - 0.5f;
  sy = fminf(fmaxf(sy, y1), y2 - 1.f);
  const float ix0f = floorf(sx), wx = sx - ix0f;
  const float iy0f = floorf(sy), wy = sy - iy0f;
  const int ix0 = (int)ix0f;
  const int ix1 = (int)fminf(ix0f + 1.f, x2 - 1.f);
  const int iy0 = (int)iy0f;
  const int iy1 = (int)fminf(iy0f + 1.f, y2 - 1.f);
  const float* img = x + (size_t)bt * 64 * 64;
  const float v00 = img[iy0 * 64 + ix0], v01 = img[iy0 * 64 + ix1];
  const float v10 = img[iy1 * 64 + ix0], v11 = img[iy1 * 64 + ix1];
  const float top = (1.f - wx) * v00 + wx * v01;
  const float bot = (1.f - wx) * v10 + wx * v11;
  crops[((size_t)(bt * 8 + r)) * 196 + p * 14 + q] =
      (_Float16)((1.f - wy) * top + wy * bot);
}

// ------------------------------------- max over T (writes 16 rows, 8-15 zero)
__global__ void k_maxT(const _Float16* __restrict__ a6g, _Float16* __restrict__ g) {
  int i = blockIdx.x * blockDim.x + threadIdx.x;
  if (i >= 16 * 32768) return;
  const int f = i & 32767, b = i >> 15;
  float m = 0.f;
  if (b < 8) {
    m = -1e30f;
    for (int t = 0; t < 30; ++t)
      m = fmaxf(m, (float)a6g[((size_t)(b * 30 + t)) * 32768 + f]);
  }
  g[(size_t)b * 32768 + f] = (_Float16)m;
}

// --------------------------------------------------------- FC as WMMA GEMM
// out[Mstore][N](f32) = A[Mpad16][K](f16) @ Wt[N][K](f16)^T + bias[N]
// A must have rows padded to a multiple of 16 (zero-filled) -> no guards.
// Inner loop double-buffered: loads for chunk k+1 in flight during WMMA k.
__global__ __launch_bounds__(32) void k_fc_wmma(
    const _Float16* __restrict__ A, const _Float16* __restrict__ Wt,
    const float* __restrict__ bias, float* __restrict__ out,
    int Mstore, int N, int K) {
  const int mbase = blockIdx.x * 16;
  const int nbase = blockIdx.y * 16;
  const int lane  = threadIdx.x;
  const int kh = lane >> 4, lid = lane & 15;
  const _Float16* __restrict__ ar = A + (size_t)(mbase + lid) * K + kh * 8;
  const _Float16* __restrict__ br = Wt + (size_t)(nbase + lid) * K + kh * 8;
  v8f c = {0.f, 0.f, 0.f, 0.f, 0.f, 0.f, 0.f, 0.f};
  h8 alo = *(const h8*)(ar);
  h8 ahi = *(const h8*)(ar + 16);
  h8 blo = *(const h8*)(br);
  h8 bhi = *(const h8*)(br + 16);
  for (int kc = 0; kc < K; kc += 32) {
    const h8 a0 = alo, a1 = ahi, b0 = blo, b1 = bhi;
    const int kn = kc + 32;
    if (kn < K) {
      alo = *(const h8*)(ar + kn);
      ahi = *(const h8*)(ar + kn + 16);
      blo = *(const h8*)(br + kn);
      bhi = *(const h8*)(br + kn + 16);
      __builtin_prefetch(ar + kn + 256, 0, 1);
      __builtin_prefetch(br + kn + 256, 0, 1);
    }
    c = __builtin_amdgcn_wmma_f32_16x16x32_f16(false, cat16(a0, a1), false, cat16(b0, b1),
                                               (short)0, c, false, false);
  }
  const int n = nbase + lid;
#pragma unroll
  for (int v = 0; v < 8; ++v) {
    const int m = mbase + v + 8 * kh;
    if (m < Mstore) out[(size_t)m * N + n] = c[v] + bias[n];
  }
}

// ------------------------------------------------------------------- GAT
template <int DIN, int DH>
__global__ void k_gat_h(const float* __restrict__ hin, const float* __restrict__ Wm,
                        const float* __restrict__ a1, const float* __restrict__ a2,
                        float* __restrict__ h, float* __restrict__ s1,
                        float* __restrict__ s2) {
  int idx = blockIdx.x * blockDim.x + threadIdx.x;  // b*240 + n
  if (idx >= 8 * 240) return;
  const float* xin = hin + (size_t)idx * DIN;
  float hv[DH];
#pragma unroll
  for (int j = 0; j < DH; ++j) hv[j] = 0.f;
  for (int k = 0; k < DIN; ++k) {
    const float xv = xin[k];
#pragma unroll
    for (int j = 0; j < DH; ++j) hv[j] += xv * Wm[k * DH + j];
  }
  float d1 = 0.f, d2 = 0.f;
#pragma unroll
  for (int j = 0; j < DH; ++j) {
    h[(size_t)idx * DH + j] = hv[j];
    d1 += hv[j] * a1[j];
    d2 += hv[j] * a2[j];
  }
  s1[idx] = d1;
  s2[idx] = d2;
}

template <int DH>
__global__ void k_gat_attn(const float* __restrict__ h, const float* __restrict__ s1,
                           const float* __restrict__ s2, float* __restrict__ out) {
  int idx = blockIdx.x * blockDim.x + threadIdx.x;  // b*240 + i
  if (idx >= 8 * 240) return;
  const int b = idx / 240, i = idx % 240;
  const int fri = i >> 3, rki = i & 7;
  const float si = s1[idx];
  float mx = -1e30f;
  for (int j = 0; j < 240; ++j) {
    const int frj = j >> 3, rkj = j & 7;
    int df = fri - frj; if (df < 0) df = -df;
    const bool adj = (fri == frj) || ((rki == rkj) && (df == 1));
    float e = si + s2[b * 240 + j];
    e = e > 0.f ? e : 0.2f * e;
    e = adj ? e : -9e15f;
    mx = fmaxf(mx, e);
  }
  float acc[DH];
#pragma unroll
  for (int d = 0; d < DH; ++d) acc[d] = 0.f;
  float den = 0.f;
  for (int j = 0; j < 240; ++j) {
    const int frj = j >> 3, rkj = j & 7;
    int df = fri - frj; if (df < 0) df = -df;
    const bool adj = (fri == frj) || ((rki == rkj) && (df == 1));
    float e = si + s2[b * 240 + j];
    e = e > 0.f ? e : 0.2f * e;
    e = adj ? e : -9e15f;
    const float w = __expf(e - mx);
    den += w;
    const float* hj = h + ((size_t)b * 240 + j) * DH;
#pragma unroll
    for (int d = 0; d < DH; ++d) acc[d] += w * hj[d];
  }
  const float inv = 1.f / den;
#pragma unroll
  for (int d = 0; d < DH; ++d) out[(size_t)idx * DH + d] = acc[d] * inv;
}

__global__ void k_elu_lsm(const float* __restrict__ in, float* __restrict__ out) {
  int idx = blockIdx.x * blockDim.x + threadIdx.x;  // 1920 rows of 16
  if (idx >= 8 * 240) return;
  float v[16];
  float mx = -1e30f;
#pragma unroll
  for (int d = 0; d < 16; ++d) {
    float z = in[(size_t)idx * 16 + d];
    z = z > 0.f ? z : (__expf(z) - 1.f);  // elu
    v[d] = z;
    mx = fmaxf(mx, z);
  }
  float s = 0.f;
#pragma unroll
  for (int d = 0; d < 16; ++d) s += __expf(v[d] - mx);
  const float lse = mx + __logf(s);
#pragma unroll
  for (int d = 0; d < 16; ++d) out[(size_t)idx * 16 + d] = v[d] - lse;
}

// -------------------------------------------------------- final feature head
__global__ void k_final(const float* __restrict__ gfc, const float* __restrict__ lfc,
                        const float* __restrict__ gatfc, float* __restrict__ out) {
  __shared__ float fc[528];
  const int b = blockIdx.x;
  for (int i = threadIdx.x; i < 256; i += blockDim.x) fc[i] = gfc[b * 256 + i];
  for (int i = threadIdx.x; i < 272; i += blockDim.x) {
    float s = 0.f;
    if (i < 256) {
      for (int n = 0; n < 240; ++n) s += lfc[((size_t)b * 240 + n) * 256 + i];
    } else {
      for (int n = 0; n < 240; ++n) s += gatfc[((size_t)b * 240 + n) * 16 + (i - 256)];
    }
    fc[256 + i] = s * (1.f / 240.f);
  }
  __syncthreads();
  for (int i = threadIdx.x; i < 1023; i += blockDim.x) {
    int nb, j;
    if (i < 528)      { nb = 1;  j = i; }
    else if (i < 792) { nb = 2;  j = i - 528; }
    else if (i < 924) { nb = 4;  j = i - 792; }
    else if (i < 990) { nb = 8;  j = i - 924; }
    else              { nb = 16; j = i - 990; }
    const int chunk = 528 / nb;
    float s = 0.f, m = -1e30f;
    for (int g = 0; g < nb; ++g) {
      const float v = fc[g * chunk + j];
      s += v;
      m = fmaxf(m, v);
    }
    out[b * 1023 + i] = s / (float)nb + m;
  }
}

// ------------------------------------------------------------------- launch
extern "C" void kernel_launch(void* const* d_in, const int* in_sizes, int n_in,
                              void* d_out, int out_size, void* d_ws, size_t ws_size,
                              hipStream_t stream) {
  (void)in_sizes; (void)n_in; (void)out_size; (void)ws_size;
  const float* x    = (const float*)d_in[0];
  const float* px   = (const float*)d_in[1];
  const float* cw[6] = {(const float*)d_in[2], (const float*)d_in[3],
                        (const float*)d_in[4], (const float*)d_in[5],
                        (const float*)d_in[6], (const float*)d_in[7]};
  const float* fcgw = (const float*)d_in[8];
  const float* fcgb = (const float*)d_in[9];
  const float* fclw = (const float*)d_in[10];
  const float* fclb = (const float*)d_in[11];
  const float* g1W  = (const float*)d_in[12];
  const float* g1a1 = (const float*)d_in[13];
  const float* g1a2 = (const float*)d_in[14];
  const float* g2W  = (const float*)d_in[15];
  const float* g2a1 = (const float*)d_in[16];
  const float* g2a2 = (const float*)d_in[17];
  float* out = (float*)d_out;

  char* ws = (char*)d_ws;
  size_t cur = 0;
  auto alloc = [&](size_t bytes) -> char* {
    char* p = ws + cur;
    cur += (bytes + 255) & ~(size_t)255;
    return p;
  };
  auto cdiv = [](long a, long b) -> unsigned { return (unsigned)((a + b - 1) / b); };

  // f16 weights, K padded to multiple of 32
  const int wK[6]  = {25, 288, 288, 576, 576, 1152};
  const int wKp[6] = {32, 288, 288, 576, 576, 1152};
  const int wO[6]  = {32, 32, 64, 64, 128, 128};
  _Float16* wh[6];
  for (int i = 0; i < 6; ++i) wh[i] = (_Float16*)alloc((size_t)wO[i] * wKp[i] * 2);
  _Float16* fcgw16 = (_Float16*)alloc((size_t)256 * 32768 * 2);
  _Float16* fclw16 = (_Float16*)alloc((size_t)256 * 1152 * 2);
  // global-branch activations (f16)
  _Float16* in0g = (_Float16*)alloc((size_t)240 * 4096 * 2);
  _Float16* a1g  = (_Float16*)alloc((size_t)240 * 32 * 4096 * 2);
  _Float16* a2g  = (_Float16*)alloc((size_t)240 * 32 * 4096 * 2);
  _Float16* p2g  = (_Float16*)alloc((size_t)240 * 32 * 1024 * 2);
  _Float16* a3g  = (_Float16*)alloc((size_t)240 * 64 * 1024 * 2);
  _Float16* a4g  = (_Float16*)alloc((size_t)240 * 64 * 1024 * 2);
  _Float16* p4g  = (_Float16*)alloc((size_t)240 * 64 * 256 * 2);
  _Float16* a5g  = (_Float16*)alloc((size_t)240 * 128 * 256 * 2);
  _Float16* a6g  = (_Float16*)alloc((size_t)240 * 128 * 256 * 2);
  _Float16* gmax = (_Float16*)alloc((size_t)16 * 32768 * 2);  // rows 8-15 zero
  _Float16* crops = (_Float16*)alloc((size_t)1920 * 196 * 2);
  // local branch aliases dead global buffers (stream-ordered reuse)
  _Float16* l1  = a1g;   // 1920*32*196 <= 240*32*4096
  _Float16* l2  = a2g;
  _Float16* p2l = p2g;   // 1920*32*49  <= 240*32*1024
  _Float16* l3  = a3g;   // 1920*64*49  <= 240*64*1024
  _Float16* l4  = a4g;
  _Float16* p4l = p4g;   // 1920*64*9   <= 240*64*256
  _Float16* l5  = a5g;   // 1920*128*9  <= 240*128*256
  _Float16* l6  = a6g;   // consumed after gmax/gfc
  // f32 buffers
  float* gfc   = (float*)alloc((size_t)8 * 256 * 4);
  float* lfc   = (float*)alloc((size_t)1920 * 256 * 4);
  float* h1    = (float*)alloc((size_t)1920 * 8 * 4);
  float* s1    = (float*)alloc((size_t)1920 * 4);
  float* s2    = (float*)alloc((size_t)1920 * 4);
  float* out1  = (float*)alloc((size_t)1920 * 8 * 4);
  float* h2    = (float*)alloc((size_t)1920 * 16 * 4);
  float* s3    = (float*)alloc((size_t)1920 * 4);
  float* s4    = (float*)alloc((size_t)1920 * 4);
  float* out2  = (float*)alloc((size_t)1920 * 16 * 4);
  float* gatfc = (float*)alloc((size_t)1920 * 16 * 4);

  // ---- weight / input conversion to f16
  for (int i = 0; i < 6; ++i) {
    const int n = wO[i] * wKp[i];
    k_pack_w<<<cdiv(n, 256), 256, 0, stream>>>(cw[i], wh[i], wO[i], wK[i], wKp[i]);
  }
  k_f32_to_f16<<<cdiv(256 * 32768, 256), 256, 0, stream>>>(fcgw, fcgw16, 256 * 32768);
  k_f32_to_f16<<<cdiv(256 * 1152, 256), 256, 0, stream>>>(fclw, fclw16, 256 * 1152);
  k_f32_to_f16<<<cdiv(240 * 4096, 256), 256, 0, stream>>>(x, in0g, 240 * 4096);

  // ---- global branch (240 images, 64x64)
  k_conv_wmma<1, 32, 5, 2, 64, 64><<<dim3(240, 32), 256, 0, stream>>>(in0g, wh[0], a1g);
  k_conv_wmma<32, 32, 3, 1, 64, 64><<<dim3(240, 32), 256, 0, stream>>>(a1g, wh[1], a2g);
  k_pool2<<<cdiv((long)240 * 32 * 32 * 32, 256), 256, 0, stream>>>(a2g, p2g, 32, 64, 64, 240);
  k_conv_wmma<32, 64, 3, 1, 32, 32><<<dim3(240, 8), 256, 0, stream>>>(p2g, wh[2], a3g);
  k_conv_wmma<64, 64, 3, 1, 32, 32><<<dim3(240, 8), 256, 0, stream>>>(a3g, wh[3], a4g);
  k_pool2<<<cdiv((long)240 * 64 * 16 * 16, 256), 256, 0, stream>>>(a4g, p4g, 64, 32, 32, 240);
  k_conv_wmma<64, 128, 3, 1, 16, 16><<<dim3(240, 2), 256, 0, stream>>>(p4g, wh[4], a5g);
  k_conv_wmma<128, 128, 3, 1, 16, 16><<<dim3(240, 2), 256, 0, stream>>>(a5g, wh[5], a6g);
  k_maxT<<<cdiv(16 * 32768, 256), 256, 0, stream>>>(a6g, gmax);
  k_fc_wmma<<<dim3(1, 16), 32, 0, stream>>>(gmax, fcgw16, fcgb, gfc, 8, 256, 32768);

  // ---- local branch (1920 crops, 14x14)
  k_roi<<<cdiv(240 * 8 * 14 * 14, 256), 256, 0, stream>>>(x, px, crops);
  k_conv_wmma<1, 32, 5, 2, 14, 14><<<dim3(1920, 2), 256, 0, stream>>>(crops, wh[0], l1);
  k_conv_wmma<32, 32, 3, 1, 14, 14><<<dim3(1920, 2), 256, 0, stream>>>(l1, wh[1], l2);
  k_pool2<<<cdiv((long)1920 * 32 * 7 * 7, 256), 256, 0, stream>>>(l2, p2l, 32, 14, 14, 1920);
  k_conv_wmma<32, 64, 3, 1, 7, 7><<<dim3(1920, 1), 256, 0, stream>>>(p2l, wh[2], l3);
  k_conv_wmma<64, 64, 3, 1, 7, 7><<<dim3(1920, 1), 256, 0, stream>>>(l3, wh[3], l4);
  k_pool2<<<cdiv((long)1920 * 64 * 3 * 3, 256), 256, 0, stream>>>(l4, p4l, 64, 7, 7, 1920);
  k_conv_wmma<64, 128, 3, 1, 3, 3><<<dim3(1920, 1), 256, 0, stream>>>(p4l, wh[4], l5);
  k_conv_wmma<128, 128, 3, 1, 3, 3><<<dim3(1920, 1), 256, 0, stream>>>(l5, wh[5], l6);
  k_fc_wmma<<<dim3(120, 16), 32, 0, stream>>>(l6, fclw16, fclb, lfc, 1920, 256, 1152);

  // ---- GAT (tiny)
  k_gat_h<256, 8><<<cdiv(1920, 128), 128, 0, stream>>>(lfc, g1W, g1a1, g1a2, h1, s1, s2);
  k_gat_attn<8><<<cdiv(1920, 128), 128, 0, stream>>>(h1, s1, s2, out1);
  k_gat_h<8, 16><<<cdiv(1920, 128), 128, 0, stream>>>(out1, g2W, g2a1, g2a2, h2, s3, s4);
  k_gat_attn<16><<<cdiv(1920, 128), 128, 0, stream>>>(h2, s3, s4, out2);
  k_elu_lsm<<<cdiv(1920, 128), 128, 0, stream>>>(out2, gatfc);

  // ---- multi-scale head
  k_final<<<8, 128, 0, stream>>>(gfc, lfc, gatfc, out);
}
